// ASTGATClassifier_44117904064512
// MI455X (gfx1250) — matrix operand order repaired and checked by reading, not compiled, not run on hardware
//
#include <hip/hip_runtime.h>
#include <hip/hip_bf16.h>
#include <math.h>

// ---------------- problem constants ----------------
#define N_NODES   50000
#define NPAD      50048      // multiple of 128 (GEMM row-block)
#define N_EDGES   400000
#define EPRIME    450000     // edges + self loops
#define N_GRAPHS  256
#define EMBD      64
#define FWID      512        // max feature width (heads*hid)
#define HEADSMAX  4
#define CHAN      128
#define GDIM      256
#define NUM_CLASSES 20
#define EPSBN     1e-5f
#define NEG_SLOPE 0.2f

typedef __attribute__((ext_vector_type(16))) _Float16 v16h;
typedef __attribute__((ext_vector_type(8)))  _Float16 v8h;
typedef __attribute__((ext_vector_type(8)))  float    v8f;
typedef __attribute__((ext_vector_type(4)))  int      v4i;

// CDNA5 async global->LDS path (ASYNCcnt-tracked), with safe fallback.
#if defined(__HIP_DEVICE_COMPILE__) && \
    __has_builtin(__builtin_amdgcn_global_load_async_to_lds_b128) && \
    __has_builtin(__builtin_amdgcn_s_wait_asynccnt)
#define USE_ASYNC_LDS 1
typedef __attribute__((address_space(1))) v4i glb_v4i;
typedef __attribute__((address_space(3))) v4i lds_v4i;
#else
#define USE_ASYNC_LDS 0
#endif

// ---------------- helpers ----------------
__device__ __forceinline__ void atomicMaxF(float* addr, float val) {
  int* ai = (int*)addr;
  int old = __float_as_int(*addr);
  while (__int_as_float(old) < val) {
    int assumed = old;
    old = atomicCAS(ai, assumed, __float_as_int(val));
    if (old == assumed) break;
  }
}

__global__ void k_fill_f32(float* p, float v, int n) {
  int i = blockIdx.x * blockDim.x + threadIdx.x;
  if (i < n) p[i] = v;
}

__global__ void k_f32_to_f16(const float* __restrict__ s, _Float16* __restrict__ d, int n) {
  int i = blockIdx.x * blockDim.x + threadIdx.x;
  if (i < n) d[i] = (_Float16)s[i];
}

// ---------------- node feature init: h = emb[x] + depth*dw + db ----------------
__global__ void k_node_init(const int* __restrict__ x, const float* __restrict__ depth,
                            const float* __restrict__ embt, const float* __restrict__ dw,
                            const float* __restrict__ db, _Float16* __restrict__ h16) {
  int i = blockIdx.x * blockDim.x + threadIdx.x;
  if (i >= N_NODES * EMBD) return;
  int node = i / EMBD, e = i % EMBD;
  float v = embt[x[node] * EMBD + e] + depth[node] * dw[e] + db[e];
  h16[(size_t)node * EMBD + e] = (_Float16)v;
}

// ---------------- WMMA GEMM: C[n][m] = sum_k A16[n][k] * W16[m][k] ----------------
// A16: [NPAD x K] f16 row-major, W16: [M x K] f16 row-major, C: [NPAD x M] f32.
// Block = 256 threads = 8 waves; block tile = 128 rows x 64 cols.
// The 64xK weight tile (rows col0..col0+63 of W are contiguous) is staged once
// into LDS (async global->LDS when available), then all 8 waves read B
// fragments from LDS while streaming A fragments from global.
__global__ __launch_bounds__(256) void k_gemm_wmma(const _Float16* __restrict__ A,
                                                   const _Float16* __restrict__ W,
                                                   float* __restrict__ C, int K, int M) {
  __shared__ __align__(16) _Float16 sB[64 * FWID];   // up to 64 KB
  const int tid  = threadIdx.x;
  const int lane = tid & 31;
  const int wave = tid >> 5;
  const int lrow = lane & 15;          // row within 16-tile (A) / col (B,D)
  const int half = lane >> 4;          // lane half selects K sub-range
  const int row0 = blockIdx.x * 128 + wave * 16;
  const int col0 = blockIdx.y * 64;

  // ---- stage B tile: 64*K halfs, contiguous region of W ----
  {
    const _Float16* gsrc = W + (size_t)col0 * K;
    const int nchunks = (64 * K) / 8;          // 16-byte chunks
#if USE_ASYNC_LDS
    for (int c = tid; c < nchunks; c += 256)
      __builtin_amdgcn_global_load_async_to_lds_b128(
          (glb_v4i*)(gsrc + c * 8), (lds_v4i*)(sB + c * 8), 0, 0);
    __builtin_amdgcn_s_wait_asynccnt(0);
    __syncthreads();
#else
    for (int c = tid; c < nchunks; c += 256)
      ((float4*)sB)[c] = ((const float4*)gsrc)[c];
    __syncthreads();
#endif
  }

  v8f acc[4] = {};
  const _Float16* arow = A + (size_t)(row0 + lrow) * K;

  for (int k0 = 0; k0 < K; k0 += 32) {
    // A fragment (16x32 f16): lanes 0-15 -> K {0..15}, lanes 16-31 -> {16..31}
    v16h afrag;
    {
      v8h lo = *(const v8h*)(arow + k0 + half * 16);
      v8h hi = *(const v8h*)(arow + k0 + half * 16 + 8);
      #pragma unroll
      for (int i = 0; i < 8; ++i) { afrag[i] = lo[i]; afrag[8 + i] = hi[i]; }
    }
    #pragma unroll
    for (int t = 0; t < 4; ++t) {
      // B fragment (32x16 f16) from LDS: lane holds column m=lrow of the tile,
      // K {0..15} (lanes 0-15) / {16..31} (lanes 16-31) contiguous.
      const _Float16* wp = sB + (size_t)(t * 16 + lrow) * K + k0 + half * 16;
      v8h lo = *(const v8h*)(wp);
      v8h hi = *(const v8h*)(wp + 8);
      v16h bfrag;
      #pragma unroll
      for (int i = 0; i < 8; ++i) { bfrag[i] = lo[i]; bfrag[8 + i] = hi[i]; }
      acc[t] = __builtin_amdgcn_wmma_f32_16x16x32_f16(
          false, afrag, false, bfrag, (short)0, acc[t], false, false);
    }
  }
  // D layout: vgpr r, lanes 0-15 -> (M=r, N=lane); lanes 16-31 -> (M=8+r, N=lane-16)
  #pragma unroll
  for (int t = 0; t < 4; ++t)
    #pragma unroll
    for (int r = 0; r < 8; ++r)
      C[(size_t)(row0 + r + 8 * half) * M + col0 + t * 16 + lrow] = acc[t][r];
}

// ---------------- attention logits: al_s/al_d [N,H] ----------------
__global__ void k_attn_logits(const float* __restrict__ xw, const float* __restrict__ asrc,
                              const float* __restrict__ adst, float* __restrict__ als,
                              float* __restrict__ ald, int H, int C) {
  int i = blockIdx.x * blockDim.x + threadIdx.x;
  if (i >= N_NODES * H) return;
  int n = i / H, h = i % H;
  const float* xp = xw + (size_t)n * H * C + h * C;
  const float* ap = asrc + h * C;
  const float* bp = adst + h * C;
  float s = 0.f, d = 0.f;
  for (int c = 0; c < C; ++c) { float v = xp[c]; s += v * ap[c]; d += v * bp[c]; }
  als[i] = s; ald[i] = d;
}

__device__ __forceinline__ void edge_sd(const int* ei, int e, int& s, int& d) {
  if (e < N_EDGES) { s = ei[e]; d = ei[N_EDGES + e]; } else { s = d = e - N_EDGES; }
}

// ---------------- edge pass 1: segment max over dst ----------------
__global__ void k_edge_max(const int* __restrict__ ei, const float* __restrict__ als,
                           const float* __restrict__ ald, float* __restrict__ segmax, int H) {
  int i = blockIdx.x * blockDim.x + threadIdx.x;
  if (i >= EPRIME * H) return;
  int e = i / H, h = i % H, s, d;
  edge_sd(ei, e, s, d);
  float v = als[s * H + h] + ald[d * H + h];
  v = v > 0.f ? v : NEG_SLOPE * v;
  atomicMaxF(&segmax[d * H + h], v);
}

// ---------------- edge pass 2: exp + denom ----------------
__global__ void k_edge_exp(const int* __restrict__ ei, const float* __restrict__ als,
                           const float* __restrict__ ald, const float* __restrict__ segmax,
                           float* __restrict__ eexp, float* __restrict__ denom, int H) {
  int i = blockIdx.x * blockDim.x + threadIdx.x;
  if (i >= EPRIME * H) return;
  int e = i / H, h = i % H, s, d;
  edge_sd(ei, e, s, d);
  float v = als[s * H + h] + ald[d * H + h];
  v = v > 0.f ? v : NEG_SLOPE * v;
  float ex = expf(v - segmax[d * H + h]);
  eexp[i] = ex;
  atomicAdd(&denom[d * H + h], ex);
}

// ---------------- edge pass 3: out[dst] += alpha * x[src] ----------------
__global__ void k_edge_aggr(const int* __restrict__ ei, const float* __restrict__ eexp,
                            const float* __restrict__ denom, const float* __restrict__ xw,
                            float* __restrict__ agg, int H, int C) {
  const int F = H * C;
  const int per = F / 4;
  long long i = (long long)blockIdx.x * blockDim.x + threadIdx.x;
  if (i >= (long long)EPRIME * per) return;
  int e = (int)(i / per);
  int f = (int)(i % per) * 4;
  int s, d;
  edge_sd(ei, e, s, d);
  int h = f / C;
  float alpha = eexp[e * H + h] / denom[d * H + h];
  float4 xv = *(const float4*)(xw + (size_t)s * F + f);
  float* o = agg + (size_t)d * F + f;
  atomicAdd(o + 0, alpha * xv.x);
  atomicAdd(o + 1, alpha * xv.y);
  atomicAdd(o + 2, alpha * xv.z);
  atomicAdd(o + 3, alpha * xv.w);
}

// ---------------- batchnorm stats: coalesced row-strip partials ----------------
// Each block scans 128 rows with lane-contiguous reads; thread t privately owns
// channels t and t+256 (if present), then 2 atomicAdds per channel.
__global__ __launch_bounds__(256) void k_bn_partial(const float* __restrict__ agg,
                                                    float* __restrict__ sum,
                                                    float* __restrict__ sq, int F) {
  int r0 = blockIdx.x * 128;
  int rend = r0 + 128; if (rend > N_NODES) rend = N_NODES;
  int f0 = threadIdx.x;
  if (f0 >= F) return;
  bool two = (f0 + 256) < F;
  float s0 = 0.f, q0 = 0.f, s1 = 0.f, q1 = 0.f;
  for (int r = r0; r < rend; ++r) {
    const float* row = agg + (size_t)r * F;
    float v = row[f0];
    s0 += v; q0 += v * v;
    if (two) { float w = row[f0 + 256]; s1 += w; q1 += w * w; }
  }
  atomicAdd(&sum[f0], s0); atomicAdd(&sq[f0], q0);
  if (two) { atomicAdd(&sum[f0 + 256], s1); atomicAdd(&sq[f0 + 256], q1); }
}

// var(x + bias) == var(x); mean shifts by the per-channel bias.
__global__ void k_bn_fin(const float* __restrict__ sum, const float* __restrict__ sq,
                         const float* __restrict__ bias, float* __restrict__ mean,
                         float* __restrict__ var, int F) {
  int f = blockIdx.x * blockDim.x + threadIdx.x;
  if (f >= F) return;
  float m = sum[f] / (float)N_NODES;
  var[f] = sq[f] / (float)N_NODES - m * m;
  mean[f] = m + bias[f];
}

// ---------------- BN + ELU apply (in place), also emit f16 for next GEMM ----------------
__global__ void k_bn_apply(float* __restrict__ agg, const float* __restrict__ bias,
                           const float* __restrict__ mean, const float* __restrict__ var,
                           const float* __restrict__ g, const float* __restrict__ be,
                           _Float16* __restrict__ h16, int F) {
  long long i = (long long)blockIdx.x * blockDim.x + threadIdx.x;
  if (i >= (long long)N_NODES * F) return;
  int f = (int)(i % F);
  float pre = agg[i] + bias[f];
  float v = (pre - mean[f]) * rsqrtf(var[f] + EPSBN) * g[f] + be[f];
  float act = v > 0.f ? v : expf(v) - 1.0f;
  agg[i] = act;
  h16[i] = (_Float16)act;
}

// ---------------- graph pooling ----------------
__global__ void k_pool(const float* __restrict__ h, const int* __restrict__ batch,
                       float* __restrict__ gsum, float* __restrict__ gmax,
                       float* __restrict__ gcnt) {
  int i = blockIdx.x * blockDim.x + threadIdx.x;
  if (i >= N_NODES * CHAN) return;
  int n = i / CHAN, f = i % CHAN;
  int g = batch[n];
  float v = h[(size_t)n * CHAN + f];
  atomicAdd(&gsum[g * CHAN + f], v);
  atomicMaxF(&gmax[g * CHAN + f], v);
  if (f == 0) atomicAdd(&gcnt[g], 1.0f);
}

__global__ void k_pool_fin(const float* __restrict__ gsum, const float* __restrict__ gmax,
                           const float* __restrict__ gcnt, float* __restrict__ emb) {
  int i = blockIdx.x * blockDim.x + threadIdx.x;
  if (i >= N_GRAPHS * CHAN) return;
  int g = i / CHAN, f = i % CHAN;
  float c = gcnt[g]; c = c > 1.f ? c : 1.f;
  emb[g * GDIM + f]        = gsum[i] / c;
  emb[g * GDIM + CHAN + f] = gmax[i];
}

// ---------------- classifier ----------------
__global__ void k_fc1(const float* __restrict__ emb, const float* __restrict__ w,
                      const float* __restrict__ b, float* __restrict__ hid) {
  int i = blockIdx.x * blockDim.x + threadIdx.x;
  if (i >= N_GRAPHS * GDIM) return;
  int g = i / GDIM, j = i % GDIM;
  const float* ep = emb + g * GDIM;
  const float* wp = w + j * GDIM;
  float s = b[j];
  for (int k = 0; k < GDIM; ++k) s += ep[k] * wp[k];
  hid[i] = s > 0.f ? s : 0.f;
}

__global__ void k_fc2(const float* __restrict__ hid, const float* __restrict__ w,
                      const float* __restrict__ b, float* __restrict__ out) {
  int i = blockIdx.x * blockDim.x + threadIdx.x;
  if (i >= N_GRAPHS * NUM_CLASSES) return;
  int g = i / NUM_CLASSES, k = i % NUM_CLASSES;
  const float* hp = hid + g * GDIM;
  const float* wp = w + k * GDIM;
  float s = b[k];
  for (int j = 0; j < GDIM; ++j) s += hp[j] * wp[j];
  out[i] = s;
}

// ---------------- host orchestration ----------------
extern "C" void kernel_launch(void* const* d_in, const int* in_sizes, int n_in,
                              void* d_out, int out_size, void* d_ws, size_t ws_size,
                              hipStream_t stream) {
  const int*   x     = (const int*)d_in[0];
  const int*   ei    = (const int*)d_in[1];
  const float* depth = (const float*)d_in[2];
  const int*   batch = (const int*)d_in[3];
  const float* embt  = (const float*)d_in[4];
  const float* dw    = (const float*)d_in[5];
  const float* db    = (const float*)d_in[6];
  const float* Ws[3]  = {(const float*)d_in[7],  (const float*)d_in[13], (const float*)d_in[19]};
  const float* ass[3] = {(const float*)d_in[8],  (const float*)d_in[14], (const float*)d_in[20]};
  const float* ads[3] = {(const float*)d_in[9],  (const float*)d_in[15], (const float*)d_in[21]};
  const float* bs[3]  = {(const float*)d_in[10], (const float*)d_in[16], (const float*)d_in[22]};
  const float* gs[3]  = {(const float*)d_in[11], (const float*)d_in[17], (const float*)d_in[23]};
  const float* bes[3] = {(const float*)d_in[12], (const float*)d_in[18], (const float*)d_in[24]};
  const float* cw1 = (const float*)d_in[25];
  const float* cb1 = (const float*)d_in[26];
  const float* cw2 = (const float*)d_in[27];
  const float* cb2 = (const float*)d_in[28];
  float* out = (float*)d_out;

  // scratch carve-out
  uint8_t* wsb = (uint8_t*)d_ws;
  size_t off = 0;
  auto alloc = [&](size_t bytes) -> void* {
    off = (off + 255) & ~(size_t)255;
    void* p = wsb + off;
    off += bytes;
    return p;
  };
  _Float16* h16 = (_Float16*)alloc((size_t)NPAD * FWID * 2);
  _Float16* w16 = (_Float16*)alloc((size_t)FWID * FWID * 2);
  float* xw     = (float*)alloc((size_t)NPAD * FWID * 4);
  float* agg    = (float*)alloc((size_t)NPAD * FWID * 4);
  float* als    = (float*)alloc((size_t)N_NODES * HEADSMAX * 4);
  float* ald    = (float*)alloc((size_t)N_NODES * HEADSMAX * 4);
  float* segmax = (float*)alloc((size_t)N_NODES * HEADSMAX * 4);
  float* denom  = (float*)alloc((size_t)N_NODES * HEADSMAX * 4);
  float* eexp   = (float*)alloc((size_t)EPRIME * HEADSMAX * 4);
  float* bsum   = (float*)alloc(FWID * 4);
  float* bsq    = (float*)alloc(FWID * 4);
  float* mean   = (float*)alloc(FWID * 4);
  float* var    = (float*)alloc(FWID * 4);
  float* gsum   = (float*)alloc(N_GRAPHS * CHAN * 4);
  float* gmax   = (float*)alloc(N_GRAPHS * CHAN * 4);
  float* gcnt   = (float*)alloc(N_GRAPHS * 4);
  float* embg   = (float*)alloc(N_GRAPHS * GDIM * 4);
  float* hid    = (float*)alloc(N_GRAPHS * GDIM * 4);

  // zero f16 activation buffer once: guarantees GEMM padding rows are zero.
  (void)hipMemsetAsync(h16, 0, (size_t)NPAD * FWID * 2, stream);

  // node features
  {
    int n = N_NODES * EMBD;
    k_node_init<<<(n + 255) / 256, 256, 0, stream>>>(x, depth, embt, dw, db, h16);
  }

  const int Ks[3] = {EMBD, FWID, FWID};
  const int Ms[3] = {FWID, FWID, CHAN};
  const int Hs[3] = {HEADSMAX, HEADSMAX, 1};
  const int Cs[3] = {CHAN, CHAN, CHAN};

  for (int L = 0; L < 3; ++L) {
    const int K = Ks[L], M = Ms[L], H = Hs[L], C = Cs[L], F = M;
    // weights -> f16
    int wn = M * K;
    k_f32_to_f16<<<(wn + 255) / 256, 256, 0, stream>>>(Ws[L], w16, wn);
    // x = h @ W^T via WMMA (LDS-staged weight tile)
    dim3 gg(NPAD / 128, M / 64);
    k_gemm_wmma<<<gg, 256, 0, stream>>>(h16, w16, xw, K, M);
    // attention logits
    int nh = N_NODES * H;
    k_attn_logits<<<(nh + 255) / 256, 256, 0, stream>>>(xw, ass[L], ads[L], als, ald, H, C);
    // segment softmax
    k_fill_f32<<<(nh + 255) / 256, 256, 0, stream>>>(segmax, -1e30f, nh);
    (void)hipMemsetAsync(denom, 0, (size_t)nh * 4, stream);
    int eh = EPRIME * H;
    k_edge_max<<<(eh + 255) / 256, 256, 0, stream>>>(ei, als, ald, segmax, H);
    k_edge_exp<<<(eh + 255) / 256, 256, 0, stream>>>(ei, als, ald, segmax, eexp, denom, H);
    // weighted aggregation
    (void)hipMemsetAsync(agg, 0, (size_t)N_NODES * F * 4, stream);
    long long tot = (long long)EPRIME * (F / 4);
    k_edge_aggr<<<(unsigned)((tot + 255) / 256), 256, 0, stream>>>(ei, eexp, denom, xw, agg, H, C);
    // batchnorm + elu (coalesced stats; apply writes agg in place + f16 for next layer)
    (void)hipMemsetAsync(bsum, 0, (size_t)F * 4, stream);
    (void)hipMemsetAsync(bsq, 0, (size_t)F * 4, stream);
    k_bn_partial<<<(N_NODES + 127) / 128, 256, 0, stream>>>(agg, bsum, bsq, F);
    k_bn_fin<<<(F + 255) / 256, 256, 0, stream>>>(bsum, bsq, bs[L], mean, var, F);
    long long nf = (long long)N_NODES * F;
    k_bn_apply<<<(unsigned)((nf + 255) / 256), 256, 0, stream>>>(agg, bs[L], mean, var,
                                                                 gs[L], bes[L], h16, F);
  }

  // pooling
  (void)hipMemsetAsync(gsum, 0, (size_t)N_GRAPHS * CHAN * 4, stream);
  (void)hipMemsetAsync(gcnt, 0, (size_t)N_GRAPHS * 4, stream);
  k_fill_f32<<<(N_GRAPHS * CHAN + 255) / 256, 256, 0, stream>>>(gmax, -1e30f, N_GRAPHS * CHAN);
  k_pool<<<(N_NODES * CHAN + 255) / 256, 256, 0, stream>>>(agg, batch, gsum, gmax, gcnt);
  k_pool_fin<<<(N_GRAPHS * CHAN + 255) / 256, 256, 0, stream>>>(gsum, gmax, gcnt, embg);

  // classifier
  k_fc1<<<(N_GRAPHS * GDIM + 255) / 256, 256, 0, stream>>>(embg, cw1, cb1, hid);
  k_fc2<<<(N_GRAPHS * NUM_CLASSES + 255) / 256, 256, 0, stream>>>(hid, cw2, cb2, out);

  (void)in_sizes; (void)n_in; (void)out_size; (void)ws_size;
}
